// ScaledMultiHeadAttention_39453569581187
// MI455X (gfx1250) — matrix operand order repaired and checked by reading, not compiled, not run on hardware
//
#include <hip/hip_runtime.h>
#include <hip/hip_bf16.h>

typedef __attribute__((ext_vector_type(16))) __bf16 v16bf;
typedef __attribute__((ext_vector_type(8)))  float  v8f;

#define D_MODEL   512
#define N_HEADS   8
#define HEAD_DIM  64
#define MAX_LEN   250
#define REL_ROWS  (2*MAX_LEN-1)   /* 499 */
#define RELP      512             /* padded rel columns */
#define BB        2
#define LL        2048
#define NTOK      (BB*LL)         /* 4096 */

// ---- native bf16 converts (gfx1250 has BF16 VALU/convert support) ----
__device__ __forceinline__ unsigned short f2bf(float f) {
    union { __bf16 h; unsigned short s; } u;
    u.h = (__bf16)f;
    return u.s;
}
__device__ __forceinline__ float bf2f(unsigned short h) {
    union { unsigned short s; __bf16 h; } u;
    u.s = h;
    return (float)u.h;
}

union Frag {
    v16bf v;
    uint4 q[2];
    unsigned u[8];
    unsigned short s[16];
};

// Load a 16-element bf16 fragment from two 16-byte spans (elements 0..7 at p0, 8..15 at p1).
__device__ __forceinline__ v16bf load_frag(const unsigned short* p0, const unsigned short* p1) {
    Frag f;
    f.q[0] = *(const uint4*)p0;
    f.q[1] = *(const uint4*)p1;
    return f.v;
}

__device__ __forceinline__ v8f wmma_bf16(v16bf a, v16bf b, v8f c) {
    return __builtin_amdgcn_wmma_f32_16x16x32_bf16(false, a, false, b, (short)0, c, false, false);
}

// ---- CDNA5 async global->LDS copy (ASYNCcnt-tracked), guarded by __has_builtin ----
#if defined(__has_builtin)
#  if __has_builtin(__builtin_amdgcn_global_load_async_to_lds_b128)
#    define HAS_ASYNC_LDS 1
#  endif
#endif
#ifndef HAS_ASYNC_LDS
#  define HAS_ASYNC_LDS 0
#endif

// Builtin signature (from hipcc diagnostic): (v4i as(1)* src, v4i as(3)* dst, Ii offset, Ii cpol)
typedef int v4i __attribute__((vector_size(16)));
typedef __attribute__((address_space(1))) v4i gv4i;
typedef __attribute__((address_space(3))) v4i lv4i;

__device__ __forceinline__ gv4i* to_glb_v4i(const void* p) {
    return (gv4i*)(unsigned long long)p;                       // inttoptr -> as(1)
}
__device__ __forceinline__ lv4i* to_lds_v4i(void* p) {
    // generic LDS pointer keeps the wave-relative offset in its low 32 bits
    return (lv4i*)(unsigned long long)(unsigned)(unsigned long long)p;  // trunc + inttoptr -> as(3)
}

__device__ __forceinline__ void async_copy_b128(const void* gsrc, void* ldst) {
#if HAS_ASYNC_LDS
    __builtin_amdgcn_global_load_async_to_lds_b128(to_glb_v4i(gsrc), to_lds_v4i(ldst), 0, 0);
#else
    *(uint4*)ldst = *(const uint4*)gsrc;
#endif
}
__device__ __forceinline__ void async_wait() {
#if HAS_ASYNC_LDS
#  if __has_builtin(__builtin_amdgcn_s_wait_asynccnt)
    __builtin_amdgcn_s_wait_asynccnt(0);
#  else
    asm volatile("s_wait_asynccnt 0x0" ::: "memory");
#  endif
#endif
}

// ---------------------------------------------------------------------------
// Generic bf16 GEMM:  C[M,N] = A[M,K] * B[K,N] (+bias)
//  - A: f32 (ABF16=0) or bf16 (ABF16=1), row-major, lda, batch stride sA (elems)
//  - B: f32. BT=0: row-major [K,N] (ldb=N). BT=1: B is [N,K] row-major (ldb=K),
//       rows >= bNv zero-padded (rel_table transpose).
//  - MODE 0: C f32 row-major (ldc) + bias
//    MODE 1: C bf16 head-major [(b*H+h), Lr, 64] split-heads (+bias)
//    MODE 2: C bf16 row-major (ldc), batch stride sC
// Block: 256 threads = 8 waves; block tile 128x128; wave tile 32x64 (2x4 WMMA tiles).
// ---------------------------------------------------------------------------
template <int ABF16, int BT, int MODE>
__global__ __launch_bounds__(256)
void gemm_bf16(const void* __restrict__ Av, long sA, int lda,
               const float* __restrict__ B, int ldb, int bNv,
               const float* __restrict__ bias,
               void* __restrict__ Cv, long sC, int ldc,
               int M, int N, int K, int Lr)
{
    __shared__ unsigned short ldsA[128 * 40];   // [row 0..127][k 0..31], pitch 40 (80B)
    __shared__ unsigned short ldsB[128 * 40];   // transposed: [n 0..127][k 0..31], pitch 40

    const int t    = threadIdx.x;
    const int lane = t & 31;
    const int wave = t >> 5;
    const int half = lane >> 4;
    const int ln   = lane & 15;
    const int m0   = blockIdx.x * 128;
    const int n0   = blockIdx.y * 128;
    const int wm   = wave >> 1;   // 0..3 -> 32 rows
    const int wn   = wave & 1;    // 0..1 -> 64 cols
    const int z    = blockIdx.z;

    v8f acc[2][4];
    #pragma unroll
    for (int tm = 0; tm < 2; ++tm)
        #pragma unroll
        for (int tn = 0; tn < 4; ++tn)
            #pragma unroll
            for (int r = 0; r < 8; ++r) acc[tm][tn][r] = 0.f;

    for (int kc = 0; kc < K; kc += 32) {
        // ---- stage A (128 x 32): each thread 16 elements of one row half ----
        {
            int row = t >> 1, hp = t & 1;
            unsigned short* dst = ldsA + row * 40 + hp * 16;
            if (ABF16) {
                const unsigned short* A = (const unsigned short*)Av + (long)z * sA;
                const unsigned short* src = A + (long)(m0 + row) * lda + kc + hp * 16;
                async_copy_b128(src, dst);
                async_copy_b128(src + 8, dst + 8);
            } else {
                const float* A = (const float*)Av + (long)z * sA;
                const float* src = A + (long)(m0 + row) * lda + kc + hp * 16;
                Frag f;
                #pragma unroll
                for (int j = 0; j < 16; ++j) f.s[j] = f2bf(src[j]);
                *(uint4*)dst       = f.q[0];
                *(uint4*)(dst + 8) = f.q[1];
            }
        }
        // ---- stage B (32 x 128) transposed into ldsB[n][k] (two 64-col halves) ----
        #pragma unroll
        for (int rep = 0; rep < 2; ++rep) {
            if (!BT) {
                int k = t >> 3, n8 = (t & 7) * 8 + rep * 64;
                const float* src = B + (long)(kc + k) * ldb + n0 + n8;
                #pragma unroll
                for (int j = 0; j < 8; ++j) ldsB[(n8 + j) * 40 + k] = f2bf(src[j]);
            } else {
                int n = (t >> 2) + rep * 64, kp = (t & 3) * 8;
                Frag f;
                if (n0 + n < bNv) {
                    const float* src = B + (long)(n0 + n) * ldb + kc + kp;
                    #pragma unroll
                    for (int j = 0; j < 8; ++j) f.s[j] = f2bf(src[j]);
                } else {
                    #pragma unroll
                    for (int j = 0; j < 8; ++j) f.s[j] = 0;
                }
                *(uint4*)(ldsB + n * 40 + kp) = f.q[0];
            }
        }
        if (ABF16) async_wait();
        __syncthreads();

        // ---- fragments + 8 WMMAs (2 A-frags x 4 B-frags) ----
        v16bf af[2], bfr[4];
        #pragma unroll
        for (int tm = 0; tm < 2; ++tm) {
            // A layout: lane ln -> M row; element e -> K = 16*(e>=8) + 8*half + (e&7)
            const unsigned short* p = ldsA + (wm * 32 + tm * 16 + ln) * 40 + 8 * half;
            af[tm] = load_frag(p, p + 16);
        }
        #pragma unroll
        for (int tn = 0; tn < 4; ++tn) {
            // B layout: lane ln -> N col; element e -> K = 16*(e>=8) + 8*half + (e&7)
            const unsigned short* p = ldsB + (wn * 64 + tn * 16 + ln) * 40 + 8 * half;
            bfr[tn] = load_frag(p, p + 16);
        }
        #pragma unroll
        for (int tm = 0; tm < 2; ++tm)
            #pragma unroll
            for (int tn = 0; tn < 4; ++tn)
                acc[tm][tn] = wmma_bf16(af[tm], bfr[tn], acc[tm][tn]);
        __syncthreads();
    }

    // ---- epilogue: C layout VGPR r -> row r + 8*half, col ln ----
    #pragma unroll
    for (int tm = 0; tm < 2; ++tm)
        #pragma unroll
        for (int tn = 0; tn < 4; ++tn)
            #pragma unroll
            for (int r = 0; r < 8; ++r) {
                int row = m0 + wm * 32 + tm * 16 + r + 8 * half;
                int col = n0 + wn * 64 + tn * 16 + ln;
                float v = acc[tm][tn][r];
                if (bias) v += bias[col];
                if (MODE == 0) {
                    float* C = (float*)Cv + (long)z * sC;
                    C[(long)row * ldc + col] = v;
                } else if (MODE == 1) {
                    unsigned short* C = (unsigned short*)Cv;
                    int H  = N >> 6;
                    int bI = row / Lr, l = row % Lr;
                    int h  = col >> 6, d = col & 63;
                    C[(((long)(bI * H + h)) * Lr + l) * 64 + d] = f2bf(v);
                } else {
                    unsigned short* C = (unsigned short*)Cv + (long)z * sC;
                    C[(long)row * ldc + col] = f2bf(v);
                }
            }
}

// ---------------------------------------------------------------------------
// Flash attention: one (b,h) per blockIdx.y; 128 query rows per block (16/wave).
// Streams keys/values in chunks of 32 (K tile staged via async global->LDS);
// online softmax; bias gathered from precomputed bias_rel; key-padding mask.
// ---------------------------------------------------------------------------
__global__ __launch_bounds__(256)
void flash_attn(const unsigned short* __restrict__ Qh,
                const unsigned short* __restrict__ Kh,
                const unsigned short* __restrict__ Vh,
                const unsigned short* __restrict__ Brel,
                const int* __restrict__ mask,
                unsigned short* __restrict__ attnb)
{
    __shared__ unsigned short ldsK[32 * 72];      // [j][d], pitch 72 (144B)
    __shared__ unsigned short ldsVt[64 * 40];     // [d][j], pitch 40
    __shared__ unsigned short ldsP[8 * 16 * 40];  // per-wave P 16x32, pitch 40

    const int t    = threadIdx.x;
    const int lane = t & 31;
    const int wave = t >> 5;
    const int half = lane >> 4;
    const int ln   = lane & 15;
    const int bh   = blockIdx.y;
    const int b    = bh >> 3, h = bh & 7;
    const int i0   = blockIdx.x * 128 + wave * 16;

    const unsigned short* Q  = Qh + (long)bh * LL * HEAD_DIM;
    const unsigned short* Kp = Kh + (long)bh * LL * HEAD_DIM;
    const unsigned short* Vp = Vh + (long)bh * LL * HEAD_DIM;
    const unsigned short* Br = Brel + (long)bh * LL * RELP;
    const int* mk = mask + b * LL;
    unsigned short* ldsPw = ldsP + wave * 16 * 40;

    // Q fragments for both K-chunks (d 0..31, 32..63), loaded once from global.
    v16bf qa[2];
    #pragma unroll
    for (int kc = 0; kc < 2; ++kc) {
        const unsigned short* p = Q + ((long)(i0 + ln)) * 64 + kc * 32 + 8 * half;
        qa[kc] = load_frag(p, p + 16);
    }

    float mrow[8], lrow[8];
    v8f o[4];
    #pragma unroll
    for (int r = 0; r < 8; ++r) { mrow[r] = -3.0e38f; lrow[r] = 0.f; }
    #pragma unroll
    for (int dt = 0; dt < 4; ++dt)
        #pragma unroll
        for (int r = 0; r < 8; ++r) o[dt][r] = 0.f;

    for (int jc = 0; jc < LL; jc += 32) {
        // ---- stage K (async direct-to-LDS) and V (transposed via VGPRs) ----
        {
            int jj = t >> 3, d8 = (t & 7) * 8;
            async_copy_b128(Kp + (long)(jc + jj) * 64 + d8, ldsK + jj * 72 + d8);
            Frag f;
            f.q[0] = *(const uint4*)(Vp + (long)(jc + jj) * 64 + d8);
            #pragma unroll
            for (int q = 0; q < 8; ++q) ldsVt[(d8 + q) * 40 + jj] = f.s[q];
        }
        async_wait();
        __syncthreads();

        // ---- S = Q * K^T (two 16x16 tiles over j), accumulated over d ----
        v8f s[2];
        #pragma unroll
        for (int jt = 0; jt < 2; ++jt) {
            #pragma unroll
            for (int r = 0; r < 8; ++r) s[jt][r] = 0.f;
            #pragma unroll
            for (int kc = 0; kc < 2; ++kc) {
                // B-frag of K^T: element e -> kdim = kc*32 + 16*(e>=8) + 8*half + (e&7), col j = jt*16 + ln
                const unsigned short* p = ldsK + (jt * 16 + ln) * 72 + kc * 32 + 8 * half;
                v16bf kb = load_frag(p, p + 16);
                s[jt] = wmma_bf16(qa[kc], kb, s[jt]);
            }
        }

        // ---- scale + relative-position bias + key mask ----
        #pragma unroll
        for (int jt = 0; jt < 2; ++jt) {
            int j  = jc + jt * 16 + ln;
            int mv = mk[j];
            #pragma unroll
            for (int r = 0; r < 8; ++r) {
                int i   = i0 + r + 8 * half;
                int rel = j - i;
                rel = rel < -(MAX_LEN - 1) ? -(MAX_LEN - 1)
                    : (rel > (MAX_LEN - 1) ? (MAX_LEN - 1) : rel);
                float bias = bf2f(Br[(long)i * RELP + rel + (MAX_LEN - 1)]);
                float v = s[jt][r] * 0.125f + bias;   // 1/sqrt(64)
                s[jt][r] = (mv == 0) ? -1.0e10f : v;
            }
        }

        // ---- online softmax (rows live in 16-lane half-groups) ----
        float alpha[8];
        #pragma unroll
        for (int r = 0; r < 8; ++r) {
            float mx = fmaxf(s[0][r], s[1][r]);
            mx = fmaxf(mx, __shfl_xor(mx, 1, 32));
            mx = fmaxf(mx, __shfl_xor(mx, 2, 32));
            mx = fmaxf(mx, __shfl_xor(mx, 4, 32));
            mx = fmaxf(mx, __shfl_xor(mx, 8, 32));
            float mnew = fmaxf(mrow[r], mx);
            float a    = __expf(mrow[r] - mnew);
            mrow[r]  = mnew;
            alpha[r] = a;
            float p0 = __expf(s[0][r] - mnew);
            float p1 = __expf(s[1][r] - mnew);
            s[0][r] = p0; s[1][r] = p1;
            float rs = p0 + p1;
            rs += __shfl_xor(rs, 1, 32);
            rs += __shfl_xor(rs, 2, 32);
            rs += __shfl_xor(rs, 4, 32);
            rs += __shfl_xor(rs, 8, 32);
            lrow[r] = lrow[r] * a + rs;
        }

        // ---- write P (C-layout -> LDS) and rescale O ----
        #pragma unroll
        for (int r = 0; r < 8; ++r) {
            int rr = r + 8 * half;
            ldsPw[rr * 40 + ln]      = f2bf(s[0][r]);
            ldsPw[rr * 40 + 16 + ln] = f2bf(s[1][r]);
            #pragma unroll
            for (int dt = 0; dt < 4; ++dt) o[dt][r] *= alpha[r];
        }

        // ---- O += P * V ----
        const unsigned short* pp = ldsPw + ln * 40 + 8 * half;
        v16bf pa = load_frag(pp, pp + 16);
        #pragma unroll
        for (int dt = 0; dt < 4; ++dt) {
            const unsigned short* p = ldsVt + (dt * 16 + ln) * 40 + 8 * half;
            v16bf vb = load_frag(p, p + 16);
            o[dt] = wmma_bf16(pa, vb, o[dt]);
        }
        __syncthreads();
    }

    // ---- normalize and store merged-head bf16 output [b, i, h*64+d] ----
    #pragma unroll
    for (int r = 0; r < 8; ++r) {
        float inv = 1.0f / lrow[r];
        int i = i0 + r + 8 * half;
        #pragma unroll
        for (int dt = 0; dt < 4; ++dt) {
            attnb[((long)(b * LL + i)) * D_MODEL + h * 64 + dt * 16 + ln] =
                f2bf(o[dt][r] * inv);
        }
    }
}

// ---------------------------------------------------------------------------
extern "C" void kernel_launch(void* const* d_in, const int* in_sizes, int n_in,
                              void* d_out, int out_size, void* d_ws, size_t ws_size,
                              hipStream_t stream)
{
    const float* query = (const float*)d_in[0];
    const float* key   = (const float*)d_in[1];
    const float* value = (const float*)d_in[2];
    const int*   mask  = (const int*)d_in[3];
    const float* Wq = (const float*)d_in[4];
    const float* bq = (const float*)d_in[5];
    const float* Wk = (const float*)d_in[6];
    const float* bk = (const float*)d_in[7];
    const float* Wv = (const float*)d_in[8];
    const float* bv = (const float*)d_in[9];
    const float* Wo = (const float*)d_in[10];
    const float* bo = (const float*)d_in[11];
    const float* rel = (const float*)d_in[12];

    char* ws = (char*)d_ws;
    unsigned short* Qh    = (unsigned short*)(ws);                 // [16][2048][64] bf16 : 4MB
    unsigned short* Kh    = (unsigned short*)(ws + (4l  << 20));   // 4MB
    unsigned short* Vh    = (unsigned short*)(ws + (8l  << 20));   // 4MB
    unsigned short* attnb = (unsigned short*)(ws + (12l << 20));   // [4096][512] bf16 : 4MB
    unsigned short* Brel  = (unsigned short*)(ws + (16l << 20));   // [16][2048][512] bf16 : 32MB

    dim3 blk(256);
    dim3 gProj(NTOK / 128, D_MODEL / 128, 1);

    // Q/K/V projections: f32 A, f32 W -> bf16 head-major
    gemm_bf16<0,0,1><<<gProj, blk, 0, stream>>>(query, 0, D_MODEL, Wq, D_MODEL, D_MODEL,
                                                bq, Qh, 0, 0, NTOK, D_MODEL, D_MODEL, LL);
    gemm_bf16<0,0,1><<<gProj, blk, 0, stream>>>(key,   0, D_MODEL, Wk, D_MODEL, D_MODEL,
                                                bk, Kh, 0, 0, NTOK, D_MODEL, D_MODEL, LL);
    gemm_bf16<0,0,1><<<gProj, blk, 0, stream>>>(value, 0, D_MODEL, Wv, D_MODEL, D_MODEL,
                                                bv, Vh, 0, 0, NTOK, D_MODEL, D_MODEL, LL);

    // bias_rel[bh,i,r] = Q[bh,i,:] . rel_table[r,:]  (batched GEMM, B transposed+padded)
    dim3 gRel(LL / 128, RELP / 128, BB * N_HEADS);
    gemm_bf16<1,1,2><<<gRel, blk, 0, stream>>>(Qh, (long)LL * HEAD_DIM, HEAD_DIM,
                                               rel, HEAD_DIM, REL_ROWS,
                                               nullptr, Brel, (long)LL * RELP, RELP,
                                               LL, RELP, HEAD_DIM, LL);

    // flash attention -> bf16 merged-head activations
    dim3 gAttn(LL / 128, BB * N_HEADS);
    flash_attn<<<gAttn, blk, 0, stream>>>(Qh, Kh, Vh, Brel, mask, attnb);

    // output projection: bf16 A, f32 Wo -> f32 d_out (+bo)
    gemm_bf16<1,0,0><<<gProj, blk, 0, stream>>>(attnb, 0, D_MODEL, Wo, D_MODEL, D_MODEL,
                                                bo, d_out, 0, D_MODEL, NTOK, D_MODEL, D_MODEL, LL);
}
